// MegBlock_76879914598799
// MI455X (gfx1250) — matrix operand order repaired and checked by reading, not compiled, not run on hardware
//
#include <hip/hip_runtime.h>
#include <hip/hip_bf16.h>

static constexpr int kH = 32;

typedef __attribute__((ext_vector_type(16))) __bf16          v16bf;
typedef __attribute__((ext_vector_type(16))) unsigned short  v16u;
typedef __attribute__((ext_vector_type(8)))  float           v8f;

union ABu { v16bf bf; v16u us; };

__device__ __forceinline__ unsigned short f32_to_bf16(float f) {
  unsigned int u = __float_as_uint(f);
  return (unsigned short)((u + 0x7FFFu + ((u >> 16) & 1u)) >> 16);
}

// softplus(x) - log(2), using raw v_exp_f32 / v_log_f32 (base-2).
// Arguments are always in the hardware-safe range: exp2 input <= 0,
// log2 input in (1,2] -> no denorm/range-reduction fixups needed.
__device__ __forceinline__ float sspf(float x) {
  const float kLog2E = 1.44269504088896340736f;
  const float kLn2   = 0.69314718055994530942f;
  float t = __builtin_amdgcn_exp2f(-fabsf(x) * kLog2E);   // exp(-|x|)
  float l = __builtin_amdgcn_logf(1.0f + t);              // log2(1+t)
  return fmaxf(x, 0.0f) + kLn2 * (l - 1.0f);
}

// Stage 16 contiguous f32 (one half-row) into LDS as bf16.
__device__ __forceinline__ void stage16(const float* __restrict__ src,
                                        unsigned short* dst) {
  const float4* s4 = (const float4*)src;
#pragma unroll
  for (int q = 0; q < 4; ++q) {
    float4 f = s4[q];
    unsigned int lo = (unsigned int)f32_to_bf16(f.x) |
                      ((unsigned int)f32_to_bf16(f.y) << 16);
    unsigned int hi = (unsigned int)f32_to_bf16(f.z) |
                      ((unsigned int)f32_to_bf16(f.w) << 16);
    uint2 p; p.x = lo; p.y = hi;
    *(uint2*)(dst + q * 4) = p;
  }
}
__device__ __forceinline__ void stage16_zero(unsigned short* dst) {
#pragma unroll
  for (int q = 0; q < 4; ++q) {
    uint2 p; p.x = 0u; p.y = 0u;
    *(uint2*)(dst + q * 4) = p;
  }
}

// One 16xKINx16 GEMM tile via v_wmma_f32_16x16x32_bf16.
// aIn: LDS row-major bf16 [16][ISTRIDE]; bp: pre-packed B operand tiles.
// A layout (16-bit 16x32): lane m=l&15, kh=l>>4; elem j: g=j>>1,p=j&1,
//   K = 2g + p + kh*16.
// B layout (16-bit 32x16): elem j: K = j + kh*16, N = l&15  (pre-packed).
template <int KIN, int ISTRIDE>
__device__ __forceinline__ v8f gemm16(const unsigned short* aIn,
                                      const unsigned short* __restrict__ bp,
                                      int nt, int ntn, int lane) {
  v8f acc = {0.f, 0.f, 0.f, 0.f, 0.f, 0.f, 0.f, 0.f};
  const int m = lane & 15;
  const int kh = lane >> 4;
#pragma unroll
  for (int kt = 0; kt < KIN / 32; ++kt) {
    ABu A;
#pragma unroll
    for (int g = 0; g < 8; ++g) {
      const int kb = kt * 32 + 2 * g + kh * 16;
      const unsigned int pair = *(const unsigned int*)(aIn + m * ISTRIDE + kb);
      A.us[2 * g]     = (unsigned short)(pair & 0xFFFFu);
      A.us[2 * g + 1] = (unsigned short)(pair >> 16);
    }
    ABu Bm;
    Bm.us = *(const v16u*)(bp + ((size_t)(kt * ntn + nt) * 32 + lane) * 16);
    acc = __builtin_amdgcn_wmma_f32_16x16x32_bf16(false, A.bf, false, Bm.bf,
                                                  (short)0, acc, false, false);
  }
  return acc;
}

// Linear + ShiftedSoftplus, result written back to LDS as bf16.
// C/D layout: acc[r] -> row m = r + kh*8, col n = nt*16 + (lane&15).
template <int KIN, int NOUT, int ISTRIDE, int OSTRIDE>
__device__ __forceinline__ void layer_ssp(const unsigned short* aIn,
                                          unsigned short* aOut,
                                          const unsigned short* __restrict__ bp,
                                          const float* __restrict__ bias,
                                          int lane) {
  const int n0 = lane & 15;
  const int kh = lane >> 4;
#pragma unroll
  for (int nt = 0; nt < NOUT / 16; ++nt) {
    v8f acc = gemm16<KIN, ISTRIDE>(aIn, bp, nt, NOUT / 16, lane);
    const float b = bias[nt * 16 + n0];
#pragma unroll
    for (int r = 0; r < 8; ++r) {
      const float y = sspf(acc[r] + b);
      aOut[(kh * 8 + r) * OSTRIDE + nt * 16 + n0] = f32_to_bf16(y);
    }
  }
  __builtin_amdgcn_wave_barrier();
}

// Final layer (NOUT=32): SSP applied, results kept in registers (f32).
template <int KIN, int ISTRIDE>
__device__ __forceinline__ void layer_out32(const unsigned short* aIn,
                                            const unsigned short* __restrict__ bp,
                                            const float* __restrict__ bias,
                                            int lane, v8f& o0, v8f& o1) {
  const int n0 = lane & 15;
#pragma unroll
  for (int nt = 0; nt < 2; ++nt) {
    v8f acc = gemm16<KIN, ISTRIDE>(aIn, bp, nt, 2, lane);
    const float b = bias[nt * 16 + n0];
    v8f o;
#pragma unroll
    for (int r = 0; r < 8; ++r) o[r] = sspf(acc[r] + b);
    if (nt == 0) o0 = o; else o1 = o;
  }
}

// ---------------- kernels ----------------

// Repack fp32 weight (Kin x Nout, row-major) into WMMA B-operand bf16 tiles.
__global__ void pack_kernel(const float* __restrict__ W,
                            unsigned short* __restrict__ out,
                            int Kin, int Nout) {
  int tid = blockIdx.x * blockDim.x + threadIdx.x;
  int total = Kin * Nout;
  if (tid >= total) return;
  int j = tid & 15;
  int lane = (tid >> 4) & 31;
  int tile = tid >> 9;
  int ntn = Nout >> 4;
  int kt = tile / ntn;
  int nt = tile - kt * ntn;
  int K = kt * 32 + j + ((lane >> 4) << 4);
  int n = nt * 16 + (lane & 15);
  out[tid] = f32_to_bf16(W[(size_t)K * Nout + n]);
}

// pre_* MLP: [32 -> 64 -> 32] with SSP, f32 output to workspace.
__global__ __launch_bounds__(128) void pre_kernel(
    const float* __restrict__ x,
    const unsigned short* __restrict__ W0, const float* __restrict__ B0,
    const unsigned short* __restrict__ W1, const float* __restrict__ B1,
    float* __restrict__ y, int rows) {
  __shared__ unsigned short sA[4][16 * 32];
  __shared__ unsigned short sT[4][16 * 64];
  const int w = threadIdx.x >> 5;
  const int lane = threadIdx.x & 31;
  const int tile = blockIdx.x * 4 + w;
  const int base = tile * 16;
  if (base >= rows) return;
  const int m = lane & 15, kh = lane >> 4;
  const int row = base + m;
  if (row < rows) stage16(x + (size_t)row * kH + kh * 16, &sA[w][m * 32 + kh * 16]);
  else            stage16_zero(&sA[w][m * 32 + kh * 16]);
  __builtin_amdgcn_wave_barrier();
  layer_ssp<32, 64, 32, 64>(sA[w], sT[w], W0, B0, lane);
  v8f o0, o1;
  layer_out32<64, 64>(sT[w], W1, B1, lane, o0, o1);
  const int n0 = lane & 15;
#pragma unroll
  for (int t = 0; t < 2; ++t) {
    const v8f& o = t ? o1 : o0;
#pragma unroll
    for (int r = 0; r < 8; ++r) {
      const int rr = base + kh * 8 + r;
      if (rr < rows) y[(size_t)rr * kH + t * 16 + n0] = o[r];
    }
  }
}

// Fused pre_e + phi_e over edges. Writes residual output, scatters e_v, e_u.
__global__ __launch_bounds__(128) void edge_kernel(
    const float* __restrict__ edges,
    const int* __restrict__ egraph,   // [E]
    const int* __restrict__ pairs,    // [E][2]
    const float* __restrict__ v_pre,
    const float* __restrict__ u_pre,
    const unsigned short* __restrict__ preW0, const float* __restrict__ preB0,
    const unsigned short* __restrict__ preW1, const float* __restrict__ preB1,
    const unsigned short* __restrict__ phW0,  const float* __restrict__ phB0,
    const unsigned short* __restrict__ phW1,  const float* __restrict__ phB1,
    const unsigned short* __restrict__ phW2,  const float* __restrict__ phB2,
    float* __restrict__ out_e,
    float* __restrict__ e_v,
    float* __restrict__ e_u,
    int E) {
  __shared__ unsigned short sA[4][16 * 128];
  __shared__ unsigned short sT[4][16 * 64];
  __shared__ unsigned short sU[4][16 * 64];
  __shared__ int sI0[4][16];
  __shared__ int sEg[4][16];
  const int w = threadIdx.x >> 5;
  const int lane = threadIdx.x & 31;
  const int tile = blockIdx.x * 4 + w;
  const int base = tile * 16;
  if (base >= E) return;
  const int m = lane & 15, kh = lane >> 4;
  int row = base + m; if (row >= E) row = E - 1;
  if (lane < 16) {
    sI0[w][m] = pairs[(size_t)row * 2];
    sEg[w][m] = egraph[row];
  }
  // raw edge features -> cols 32..63 (pre_e input, computed in place)
  stage16(edges + (size_t)row * kH + kh * 16, &sA[w][m * 128 + 32 + kh * 16]);
  __builtin_amdgcn_wave_barrier();
  const int i0 = sI0[w][m];
  const int i1 = pairs[(size_t)row * 2 + 1];
  const int g  = sEg[w][m];
  stage16(v_pre + (size_t)i0 * kH + kh * 16, &sA[w][m * 128 + 0  + kh * 16]);
  stage16(v_pre + (size_t)i1 * kH + kh * 16, &sA[w][m * 128 + 64 + kh * 16]);
  stage16(u_pre + (size_t)g  * kH + kh * 16, &sA[w][m * 128 + 96 + kh * 16]);
  __builtin_amdgcn_wave_barrier();
  // pre_e: 32 -> 64 -> 32, written back into cols 32..63
  layer_ssp<32, 64, 128, 64>(&sA[w][32], sT[w], preW0, preB0, lane);
  layer_ssp<64, 32, 64, 128>(sT[w], &sA[w][32], preW1, preB1, lane);
  // phi_e: 128 -> 64 -> 64 -> 32
  layer_ssp<128, 64, 128, 64>(sA[w], sT[w], phW0, phB0, lane);
  layer_ssp<64, 64, 64, 64>(sT[w], sU[w], phW1, phB1, lane);
  v8f o0, o1;
  layer_out32<64, 64>(sU[w], phW2, phB2, lane, o0, o1);
  const int n0 = lane & 15;
  const int gu = sEg[w][0];  // edge segments are 16-aligned -> tile-uniform
#pragma unroll
  for (int t = 0; t < 2; ++t) {
    const v8f& o = t ? o1 : o0;
    float csum = 0.0f;
#pragma unroll
    for (int r = 0; r < 8; ++r) {
      const int mm = kh * 8 + r;
      const int rr = base + mm;
      const int n = t * 16 + n0;
      const float ep = o[r];
      if (rr < E) {
        out_e[(size_t)rr * kH + n] = edges[(size_t)rr * kH + n] + ep;
        atomicAdd(&e_v[(size_t)sI0[w][mm] * kH + n], ep);
        csum += ep;
      }
    }
    csum += __shfl_xor(csum, 16, 32);
    if (kh == 0) atomicAdd(&e_u[(size_t)gu * kH + t * 16 + n0], csum);
  }
}

// phi_v over nodes: [e_v | v | u_v] -> 96 -> 64 -> 64 -> 32.
__global__ __launch_bounds__(128) void node_kernel(
    const float* __restrict__ nodes,
    const int* __restrict__ nidx,
    const float* __restrict__ e_v,
    const float* __restrict__ v_pre,
    const float* __restrict__ u_pre,
    const unsigned short* __restrict__ W0, const float* __restrict__ B0,
    const unsigned short* __restrict__ W1, const float* __restrict__ B1,
    const unsigned short* __restrict__ W2, const float* __restrict__ B2,
    float* __restrict__ out_n,
    float* __restrict__ v_u,
    int N, int add) {
  __shared__ unsigned short sA[4][16 * 96];
  __shared__ unsigned short sT[4][16 * 64];
  __shared__ unsigned short sU[4][16 * 64];
  __shared__ int sG[4][16];
  const int w = threadIdx.x >> 5;
  const int lane = threadIdx.x & 31;
  const int tile = blockIdx.x * 4 + w;
  const int base = tile * 16;
  if (base >= N) return;
  const int m = lane & 15, kh = lane >> 4;
  int row = base + m; if (row >= N) row = N - 1;
  if (lane < 16) sG[w][m] = nidx[row];
  stage16(e_v   + (size_t)row * kH + kh * 16, &sA[w][m * 96 + 0  + kh * 16]);
  stage16(v_pre + (size_t)row * kH + kh * 16, &sA[w][m * 96 + 32 + kh * 16]);
  __builtin_amdgcn_wave_barrier();
  const int g = sG[w][m];
  stage16(u_pre + (size_t)g * kH + kh * 16, &sA[w][m * 96 + 64 + kh * 16]);
  __builtin_amdgcn_wave_barrier();
  layer_ssp<96, 64, 96, 64>(sA[w], sT[w], W0, B0, lane);
  layer_ssp<64, 64, 64, 64>(sT[w], sU[w], W1, B1, lane);
  v8f o0, o1;
  layer_out32<64, 64>(sU[w], W2, B2, lane, o0, o1);
  const int n0 = lane & 15;
#pragma unroll
  for (int t = 0; t < 2; ++t) {
    const v8f& o = t ? o1 : o0;
#pragma unroll
    for (int r = 0; r < 8; ++r) {
      const int mm = kh * 8 + r;
      const int rr = base + mm;
      if (rr >= N) continue;
      const int n = t * 16 + n0;
      const float vp = o[r];
      if (add) out_n[(size_t)rr * kH + n] += vp;
      else     out_n[(size_t)rr * kH + n] = nodes[(size_t)rr * kH + n] + vp;
      atomicAdd(&v_u[(size_t)sG[w][mm] * kH + n], vp);
    }
  }
}

// phi_u over graphs: [e_u | v_u | u] -> 96 -> 64 -> 64 -> 32.
__global__ __launch_bounds__(128) void state_kernel(
    const float* __restrict__ state,
    const float* __restrict__ e_u,
    const float* __restrict__ v_u,
    const float* __restrict__ u_pre,
    const unsigned short* __restrict__ W0, const float* __restrict__ B0,
    const unsigned short* __restrict__ W1, const float* __restrict__ B1,
    const unsigned short* __restrict__ W2, const float* __restrict__ B2,
    float* __restrict__ out_s, int Bn, int add) {
  __shared__ unsigned short sA[4][16 * 96];
  __shared__ unsigned short sT[4][16 * 64];
  __shared__ unsigned short sU[4][16 * 64];
  const int w = threadIdx.x >> 5;
  const int lane = threadIdx.x & 31;
  const int tile = blockIdx.x * 4 + w;
  const int base = tile * 16;
  if (base >= Bn) return;
  const int m = lane & 15, kh = lane >> 4;
  const int row = base + m;
  if (row < Bn) {
    stage16(e_u   + (size_t)row * kH + kh * 16, &sA[w][m * 96 + 0  + kh * 16]);
    stage16(v_u   + (size_t)row * kH + kh * 16, &sA[w][m * 96 + 32 + kh * 16]);
    stage16(u_pre + (size_t)row * kH + kh * 16, &sA[w][m * 96 + 64 + kh * 16]);
  } else {
    stage16_zero(&sA[w][m * 96 + 0  + kh * 16]);
    stage16_zero(&sA[w][m * 96 + 32 + kh * 16]);
    stage16_zero(&sA[w][m * 96 + 64 + kh * 16]);
  }
  __builtin_amdgcn_wave_barrier();
  layer_ssp<96, 64, 96, 64>(sA[w], sT[w], W0, B0, lane);
  layer_ssp<64, 64, 64, 64>(sT[w], sU[w], W1, B1, lane);
  v8f o0, o1;
  layer_out32<64, 64>(sU[w], W2, B2, lane, o0, o1);
  const int n0 = lane & 15;
#pragma unroll
  for (int t = 0; t < 2; ++t) {
    const v8f& o = t ? o1 : o0;
#pragma unroll
    for (int r = 0; r < 8; ++r) {
      const int rr = base + kh * 8 + r;
      if (rr >= Bn) continue;
      const int n = t * 16 + n0;
      if (add) out_s[(size_t)rr * kH + n] += o[r];
      else     out_s[(size_t)rr * kH + n] = state[(size_t)rr * kH + n] + o[r];
    }
  }
}

// ---------------- host launcher ----------------

extern "C" void kernel_launch(void* const* d_in, const int* in_sizes, int n_in,
                              void* d_out, int out_size, void* d_ws, size_t ws_size,
                              hipStream_t stream) {
  (void)n_in; (void)out_size; (void)ws_size;
  const int B  = in_sizes[1];
  const int N  = in_sizes[2];
  const int EC = in_sizes[6];
  const int EV = in_sizes[11];

  const float* nodes      = (const float*)d_in[0];
  const int*   node_index = (const int*)d_in[2];
  const float* state      = (const float*)d_in[3];
  const int*   che_eg     = (const int*)d_in[6];
  const int*   che_pairs  = (const int*)d_in[7];
  const float* che_edges  = (const float*)d_in[8];
  const int*   vdw_eg     = (const int*)d_in[11];
  const int*   vdw_pairs  = (const int*)d_in[12];
  const float* vdw_edges  = (const float*)d_in[13];

  // workspace carve-out
  char* ws = (char*)d_ws;
  size_t off = 0;
  auto carve = [&](size_t bytes) -> char* {
    char* p = ws + off;
    off += (bytes + 255) & ~(size_t)255;
    return p;
  };
  float* v_pre  = (float*)carve((size_t)N * kH * 4);
  float* u_pre  = (float*)carve((size_t)B * kH * 4);
  char* zstart  = ws + off;
  float* e_v_che = (float*)carve((size_t)N * kH * 4);
  float* e_v_vdw = (float*)carve((size_t)N * kH * 4);
  float* e_u_che = (float*)carve((size_t)B * kH * 4);
  float* e_u_vdw = (float*)carve((size_t)B * kH * 4);
  float* v_u_che = (float*)carve((size_t)B * kH * 4);
  float* v_u_vdw = (float*)carve((size_t)B * kH * 4);
  size_t zbytes = (size_t)((ws + off) - zstart);

  // params (d_in index of each W; biases are W-index + 1)
  static const int widx[26]  = {14,16, 18,20, 22,24, 26,28,
                                30,32,34, 36,38,40,
                                42,44,46, 48,50,52,
                                54,56,58, 60,62,64};
  static const int wkin[26]  = {32,64, 32,64, 32,64, 32,64,
                                128,64,64, 128,64,64,
                                96,64,64, 96,64,64,
                                96,64,64, 96,64,64};
  static const int wnout[26] = {64,32, 64,32, 64,32, 64,32,
                                64,64,32, 64,64,32,
                                64,64,32, 64,64,32,
                                64,64,32, 64,64,32};
  unsigned short* pk[26];
  for (int i = 0; i < 26; ++i)
    pk[i] = (unsigned short*)carve((size_t)wkin[i] * wnout[i] * 2);

  hipMemsetAsync(zstart, 0, zbytes, stream);
  for (int i = 0; i < 26; ++i) {
    int total = wkin[i] * wnout[i];
    pack_kernel<<<(total + 255) / 256, 256, 0, stream>>>(
        (const float*)d_in[widx[i]], pk[i], wkin[i], wnout[i]);
  }

  float* out_che = (float*)d_out;
  float* out_vdw = out_che + (size_t)EC * kH;
  float* out_n   = out_vdw + (size_t)EV * kH;
  float* out_s   = out_n + (size_t)N * kH;

  // pre_v, pre_u
  {
    int tiles = (N + 15) / 16;
    pre_kernel<<<(tiles + 3) / 4, 128, 0, stream>>>(
        nodes, pk[4], (const float*)d_in[23], pk[5], (const float*)d_in[25],
        v_pre, N);
  }
  {
    int tiles = (B + 15) / 16;
    pre_kernel<<<(tiles + 3) / 4, 128, 0, stream>>>(
        state, pk[6], (const float*)d_in[27], pk[7], (const float*)d_in[29],
        u_pre, B);
  }

  // edges (fused pre_e + phi_e)
  {
    int tiles = (EC + 15) / 16;
    edge_kernel<<<(tiles + 3) / 4, 128, 0, stream>>>(
        che_edges, che_eg, che_pairs, v_pre, u_pre,
        pk[0], (const float*)d_in[15], pk[1], (const float*)d_in[17],
        pk[8], (const float*)d_in[31], pk[9], (const float*)d_in[33],
        pk[10], (const float*)d_in[35],
        out_che, e_v_che, e_u_che, EC);
  }
  {
    int tiles = (EV + 15) / 16;
    edge_kernel<<<(tiles + 3) / 4, 128, 0, stream>>>(
        vdw_edges, vdw_eg, vdw_pairs, v_pre, u_pre,
        pk[2], (const float*)d_in[19], pk[3], (const float*)d_in[21],
        pk[11], (const float*)d_in[37], pk[12], (const float*)d_in[39],
        pk[13], (const float*)d_in[41],
        out_vdw, e_v_vdw, e_u_vdw, EV);
  }

  // nodes (phi_v, both branches; second accumulates)
  {
    int tiles = (N + 15) / 16;
    node_kernel<<<(tiles + 3) / 4, 128, 0, stream>>>(
        nodes, node_index, e_v_che, v_pre, u_pre,
        pk[14], (const float*)d_in[43], pk[15], (const float*)d_in[45],
        pk[16], (const float*)d_in[47],
        out_n, v_u_che, N, 0);
    node_kernel<<<(tiles + 3) / 4, 128, 0, stream>>>(
        nodes, node_index, e_v_vdw, v_pre, u_pre,
        pk[17], (const float*)d_in[49], pk[18], (const float*)d_in[51],
        pk[19], (const float*)d_in[53],
        out_n, v_u_vdw, N, 1);
  }

  // state (phi_u, both branches)
  {
    int tiles = (B + 15) / 16;
    state_kernel<<<(tiles + 3) / 4, 128, 0, stream>>>(
        state, e_u_che, v_u_che, u_pre,
        pk[20], (const float*)d_in[55], pk[21], (const float*)d_in[57],
        pk[22], (const float*)d_in[59],
        out_s, B, 0);
    state_kernel<<<(tiles + 3) / 4, 128, 0, stream>>>(
        state, e_u_vdw, v_u_vdw, u_pre,
        pk[23], (const float*)d_in[61], pk[24], (const float*)d_in[63],
        pk[25], (const float*)d_in[65],
        out_s, B, 1);
  }
}
